// BitFeedForward_50276887167090
// MI455X (gfx1250) — 8-bit matrix operand order repaired and checked by reading, compiled, not run on hardware
//
#include <hip/hip_runtime.h>
#include <cstdint>
#include <cstddef>

#define QEPS 1e-5f

typedef __attribute__((ext_vector_type(8))) int v8i;

__device__ __forceinline__ void async_load_16B(const void* gptr, void* lptr) {
  unsigned lds = (unsigned)(unsigned long long)lptr;
  asm volatile("global_load_async_to_lds_b128 %0, %1, off"
               :: "v"(lds), "v"(gptr) : "memory");
}
__device__ __forceinline__ void wait_async0() {
  asm volatile("s_wait_asynccnt 0x0" ::: "memory");
}

__global__ __launch_bounds__(256) void abs_sum_part_kernel(
    const float4* __restrict__ w4, long long n4, float* __restrict__ part) {
  __shared__ float red[256];
  float acc = 0.f;
  for (long long i = (long long)blockIdx.x * 256 + threadIdx.x; i < n4;
       i += (long long)gridDim.x * 256) {
    float4 v = w4[i];
    acc += fabsf(v.x) + fabsf(v.y) + fabsf(v.z) + fabsf(v.w);
  }
  red[threadIdx.x] = acc;
  __syncthreads();
  for (int s = 128; s > 0; s >>= 1) {
    if ((int)threadIdx.x < s) red[threadIdx.x] += red[threadIdx.x + s];
    __syncthreads();
  }
  if (threadIdx.x == 0) part[blockIdx.x] = red[0];
}

__global__ __launch_bounds__(256) void reduce_final_kernel(
    const float* __restrict__ part, float* __restrict__ out) {
  __shared__ float red[256];
  float acc = 0.f;
  for (int i = threadIdx.x; i < 1024; i += 256) acc += part[i];
  red[threadIdx.x] = acc;
  __syncthreads();
  for (int s = 128; s > 0; s >>= 1) {
    if ((int)threadIdx.x < s) red[threadIdx.x] += red[threadIdx.x + s];
    __syncthreads();
  }
  if (threadIdx.x == 0) out[0] = red[0];
}

__device__ __forceinline__ signed char quant1(float x, float scale, float lo,
                                              float hi) {
  float q = rintf(x * scale);
  return (signed char)fminf(fmaxf(q, lo), hi);
}

__global__ __launch_bounds__(256) void wquant_kernel(
    const float4* __restrict__ w4, char4* __restrict__ wq4, long long n4,
    const float* __restrict__ sum, float inv_n) {
  const float scale = 1.0f / fmaxf(sum[0] * inv_n, QEPS);
  long long i = (long long)blockIdx.x * 256 + threadIdx.x;
  if (i < n4) {
    float4 v = w4[i];
    char4 q;
    q.x = quant1(v.x, scale, -1.f, 1.f);
    q.y = quant1(v.y, scale, -1.f, 1.f);
    q.z = quant1(v.z, scale, -1.f, 1.f);
    q.w = quant1(v.w, scale, -1.f, 1.f);
    wq4[i] = q;
  }
}

__global__ __launch_bounds__(256) void aquant_kernel(
    const float* __restrict__ x, int8_t* __restrict__ xq,
    float* __restrict__ row_dq, int K) {
  __shared__ float red[256];
  const float4* row4 = (const float4*)(x + (size_t)blockIdx.x * K);
  char4* qrow4 = (char4*)(xq + (size_t)blockIdx.x * K);
  const int K4 = K >> 2;
  float mx = 0.f;
  for (int i = threadIdx.x; i < K4; i += 256) {
    float4 v = row4[i];
    mx = fmaxf(mx, fmaxf(fmaxf(fabsf(v.x), fabsf(v.y)),
                         fmaxf(fabsf(v.z), fabsf(v.w))));
  }
  red[threadIdx.x] = mx;
  __syncthreads();
  for (int s = 128; s > 0; s >>= 1) {
    if ((int)threadIdx.x < s)
      red[threadIdx.x] = fmaxf(red[threadIdx.x], red[threadIdx.x + s]);
    __syncthreads();
  }
  const float m = fmaxf(red[0], QEPS);
  const float scale = 127.f / m;
  for (int i = threadIdx.x; i < K4; i += 256) {
    float4 v = row4[i];
    char4 q;
    q.x = quant1(v.x, scale, -128.f, 127.f);
    q.y = quant1(v.y, scale, -128.f, 127.f);
    q.z = quant1(v.z, scale, -128.f, 127.f);
    q.w = quant1(v.w, scale, -128.f, 127.f);
    qrow4[i] = q;
  }
  if (threadIdx.x == 0) row_dq[blockIdx.x] = m / 127.f;
}

__device__ __forceinline__ int a_dw(int v, int h) {
  return (v & 3) + (h << 2) + ((v >> 2) << 3);
}
__device__ __forceinline__ int b_dw(int v, int h) {
  return (v & 3) + (h << 2) + ((v >> 2) << 3);
}

__device__ __forceinline__ void compute_step(const int* __restrict__ lA,
                                             const int* __restrict__ lB,
                                             int sub, int wy, int wx, int l16,
                                             int h, v8i (*acc)[4]) {
  const int kofs = sub << 4;
  v8i a[2];
#pragma unroll
  for (int mt = 0; mt < 2; ++mt) {
    const int arow = (wy << 5) + (mt << 4) + l16;
#pragma unroll
    for (int v = 0; v < 8; ++v) a[mt][v] = lA[arow * 36 + kofs + a_dw(v, h)];
  }
  v8i b[2];
#pragma unroll
  for (int v = 0; v < 8; ++v)
    b[0][v] = lB[((wx << 6) + l16) * 36 + kofs + b_dw(v, h)];
#pragma unroll
  for (int nt = 0; nt < 4; ++nt) {
    if (nt < 3) {
      const int brow = (wx << 6) + ((nt + 1) << 4) + l16;
#pragma unroll
      for (int v = 0; v < 8; ++v)
        b[(nt + 1) & 1][v] = lB[brow * 36 + kofs + b_dw(v, h)];
    }
#pragma unroll
    for (int mt = 0; mt < 2; ++mt)
      acc[mt][nt] = __builtin_amdgcn_wmma_i32_16x16x64_iu8(
          true, a[mt], true, b[nt & 1], acc[mt][nt], false, false);
  }
  __builtin_amdgcn_sched_group_barrier(0x100, 6, 0);
#pragma unroll
  for (int g = 0; g < 3; ++g) {
    __builtin_amdgcn_sched_group_barrier(0x008, 2, 0);
    __builtin_amdgcn_sched_group_barrier(0x100, 2, 0);
  }
  __builtin_amdgcn_sched_group_barrier(0x008, 2, 0);
}

__global__ __launch_bounds__(256) void gemm_i8_wmma_kernel(
    const int8_t* __restrict__ A, const int8_t* __restrict__ B,
    float* __restrict__ C, const float* __restrict__ row_dq,
    const float* __restrict__ wsum, float inv_wn,
    int M, int N, int K, int do_relu) {
  __shared__ __align__(16) int lds_a[2][128 * 36];
  __shared__ __align__(16) int lds_b[2][128 * 36];

  const int tid = threadIdx.x;
  const int wave = tid >> 5;
  const int lane = tid & 31;
  const int l16 = lane & 15;
  const int h = lane >> 4;
  const int wy = wave >> 1;
  const int wx = wave & 1;

  const int m0 = blockIdx.y * 128;
  const int n0 = blockIdx.x * 128;

  v8i acc[2][4] = {};

  auto issue_tile = [&](int kb, int buf) {
#pragma unroll
    for (int i = 0; i < 4; ++i) {
      const int c = tid + (i << 8);
      const int row = c >> 3, col = c & 7;
      async_load_16B(A + (size_t)(m0 + row) * K + kb + col * 16,
                     &lds_a[buf][row * 36 + col * 4]);
      async_load_16B(B + (size_t)(n0 + row) * K + kb + col * 16,
                     &lds_b[buf][row * 36 + col * 4]);
    }
  };

  const int ksteps = K >> 7;
  issue_tile(0, 0);
  for (int ks = 0; ks < ksteps; ++ks) {
    wait_async0();
    __syncthreads();
    if (ks + 1 < ksteps) {
      issue_tile((ks + 1) << 7, (ks + 1) & 1);
      if (ks + 2 < ksteps) {
        __builtin_prefetch(A + (size_t)(m0 + (tid >> 3)) * K + ((ks + 2) << 7),
                           0, 1);
        __builtin_prefetch(B + (size_t)(n0 + (tid >> 3)) * K + ((ks + 2) << 7),
                           0, 1);
      }
    }
    const int* lA = lds_a[ks & 1];
    const int* lB = lds_b[ks & 1];
    compute_step(lA, lB, 0, wy, wx, l16, h, acc);
    compute_step(lA, lB, 1, wy, wx, l16, h, acc);
  }

  const float wdq = fmaxf(wsum[0] * inv_wn, QEPS);
  float rdq[2][8];
#pragma unroll
  for (int mt = 0; mt < 2; ++mt)
#pragma unroll
    for (int v = 0; v < 8; ++v)
      rdq[mt][v] =
          row_dq[m0 + (wy << 5) + (mt << 4) + (h << 3) + v] * wdq;
#pragma unroll
  for (int nt = 0; nt < 4; ++nt) {
#pragma unroll
    for (int mt = 0; mt < 2; ++mt) {
#pragma unroll
      for (int v = 0; v < 8; ++v) {
        const int m = m0 + (wy << 5) + (mt << 4) + (h << 3) + v;
        const int n = n0 + (wx << 6) + (nt << 4) + l16;
        float val = (float)acc[mt][nt][v] * rdq[mt][v];
        if (do_relu) val = fmaxf(val, 0.f);
        C[(size_t)m * N + n] = val;
      }
    }
  }
}

extern "C" void kernel_launch(void* const* d_in, const int* in_sizes, int n_in,
                              void* d_out, int out_size, void* d_ws,
                              size_t ws_size, hipStream_t stream) {
  (void)in_sizes; (void)n_in; (void)out_size; (void)ws_size;
  const float* x  = (const float*)d_in[0];
  const float* w1 = (const float*)d_in[1];
  const float* w2 = (const float*)d_in[2];
  float* out = (float*)d_out;

  const long long M = 8192, D = 2048, I = 8192;
  const long long n_w = D * I;

  char* ws = (char*)d_ws;
  float* hdr   = (float*)ws;
  float* part1 = hdr + 256;
  float* part2 = hdr + 1280;
  int8_t* w1q = (int8_t*)(ws + 16384);
  int8_t* w2q = w1q + n_w;
  int8_t* xq  = w2q + n_w;
  float*  xdq = (float*)(xq + M * D);
  float*  hdq = xdq + M;
  int8_t* hq  = (int8_t*)(hdq + M);
  float*  h   = (float*)(hq + M * I);

  abs_sum_part_kernel<<<1024, 256, 0, stream>>>((const float4*)w1, n_w / 4,
                                                part1);
  reduce_final_kernel<<<1, 256, 0, stream>>>(part1, &hdr[0]);
  abs_sum_part_kernel<<<1024, 256, 0, stream>>>((const float4*)w2, n_w / 4,
                                                part2);
  reduce_final_kernel<<<1, 256, 0, stream>>>(part2, &hdr[1]);

  wquant_kernel<<<(unsigned)((n_w / 4 + 255) / 256), 256, 0, stream>>>(
      (const float4*)w1, (char4*)w1q, n_w / 4, &hdr[0], 1.0f / (float)n_w);
  wquant_kernel<<<(unsigned)((n_w / 4 + 255) / 256), 256, 0, stream>>>(
      (const float4*)w2, (char4*)w2q, n_w / 4, &hdr[1], 1.0f / (float)n_w);

  aquant_kernel<<<(unsigned)M, 256, 0, stream>>>(x, xq, xdq, (int)D);

  gemm_i8_wmma_kernel<<<dim3((unsigned)(I / 128), (unsigned)(M / 128)), 256, 0,
                        stream>>>(xq, w1q, h, xdq, &hdr[0],
                                  1.0f / (float)n_w, (int)M, (int)I, (int)D, 1);

  aquant_kernel<<<(unsigned)M, 256, 0, stream>>>(h, hq, hdq, (int)I);

  gemm_i8_wmma_kernel<<<dim3((unsigned)(D / 128), (unsigned)(M / 128)), 256, 0,
                        stream>>>(hq, w2q, out, hdq, &hdr[1],
                                  1.0f / (float)n_w, (int)M, (int)D, (int)I, 0);
}
